// EMLComposition_79663053406677
// MI455X (gfx1250) — matrix operand order repaired and checked by reading, not compiled, not run on hardware
//
#include <hip/hip_runtime.h>
#include <hip/hip_bf16.h>
#include <stdint.h>

typedef __attribute__((ext_vector_type(16))) _Float16 v16h;
typedef __attribute__((ext_vector_type(8)))  float    v8f;

union AF { v16h v; uint32_t u[8]; };

#define NB      16
#define CCH     256
#define HID     512
#define NPIX    65536      // B * 4096
#define NGRP    16384      // B * 1024

__device__ inline float gelu_exact(float x) {
    return 0.5f * x * (1.0f + erff(x * 0.70710678118654752f));
}

__device__ inline float wave_sum32(float v) {
    #pragma unroll
    for (int m = 16; m >= 1; m >>= 1) v += __shfl_xor(v, m, 32);
    return v;
}

// ---------------------------------------------------------------------------
// Pack an f32 KxN weight matrix into CDNA5 WMMA B-fragment f16 tiles.
// dst[((kt*ntiles + nt)*32 + lane)*8 + j] packs K = kt*32 + (lane>=16?16:0) + 2j (+1)
// for column N = nt*16 + (lane&15). One contiguous 32B v16h load per lane.
// ---------------------------------------------------------------------------
__global__ void pack_b_kernel(const float* __restrict__ src, int N, int ntiles,
                              uint32_t* __restrict__ dst, int total) {
    int id = blockIdx.x * blockDim.x + threadIdx.x;
    if (id >= total) return;
    int lane = id & 31;
    int tile = id >> 5;
    int nt = tile % ntiles;
    int kt = tile / ntiles;
    int n = nt * 16 + (lane & 15);
    int kbase = kt * 32 + (lane >> 4) * 16;
    #pragma unroll
    for (int j = 0; j < 8; ++j) {
        int k = kbase + 2 * j;
        _Float16 h0 = (_Float16)src[(size_t)k * N + n];
        _Float16 h1 = (_Float16)src[(size_t)(k + 1) * N + n];
        union { _Float16 h; unsigned short u; } c0, c1;
        c0.h = h0; c1.h = h1;
        dst[(size_t)id * 8 + j] = (uint32_t)c0.u | ((uint32_t)c1.u << 16);
    }
}

// ---------------------------------------------------------------------------
// Stage 1: per 2x2 group: LayerNorm each pixel, group mean, per-pixel variance.
// Block = 128 threads (4 waves; wave g handles group element g).
// ---------------------------------------------------------------------------
__global__ __launch_bounds__(128) void stage1_kernel(
    const float* __restrict__ states, const float* __restrict__ lng,
    const float* __restrict__ lnb, _Float16* __restrict__ normF,
    _Float16* __restrict__ gmF, float* __restrict__ varOut) {
    __shared__ float norm_s[4 * CCH];
    const int lane = threadIdx.x & 31;
    const int wid  = threadIdx.x >> 5;          // group element g = ry*2+rx
    const int gb   = blockIdx.x;                // global group index
    const int b    = gb >> 10;
    const int patch = gb & 1023;
    const int py = patch >> 5, px = patch & 31;
    const int ry = wid >> 1, rx = wid & 1;
    const int spatial = (py * 2 + ry) * 64 + (px * 2 + rx);
    const float* x = states + ((size_t)b * 4096 + spatial) * CCH;

    float xs[8];
    #pragma unroll
    for (int i = 0; i < 8; ++i) xs[i] = x[lane + i * 32];
    float s = 0.f;
    #pragma unroll
    for (int i = 0; i < 8; ++i) s += xs[i];
    s = wave_sum32(s);
    const float mean = s * (1.0f / 256.0f);
    float v = 0.f;
    #pragma unroll
    for (int i = 0; i < 8; ++i) { float d = xs[i] - mean; v += d * d; }
    v = wave_sum32(v);
    const float rstd = rsqrtf(v * (1.0f / 256.0f) + 1e-5f);

    float ns[8];
    #pragma unroll
    for (int i = 0; i < 8; ++i) {
        int c = lane + i * 32;
        ns[i] = (xs[i] - mean) * rstd * lng[c] + lnb[c];
        norm_s[wid * CCH + c] = ns[i];
    }
    __syncthreads();

    const size_t P = (size_t)gb * 4 + wid;
    float rv = 0.f;
    #pragma unroll
    for (int i = 0; i < 8; ++i) {
        int c = lane + i * 32;
        float gm = 0.25f * (norm_s[c] + norm_s[CCH + c] + norm_s[2 * CCH + c] + norm_s[3 * CCH + c]);
        normF[P * CCH + c] = (_Float16)ns[i];
        if (wid == 0) gmF[(size_t)gb * CCH + c] = (_Float16)gm;
        float d = ns[i] - gm;
        rv += d * d;
    }
    rv = wave_sum32(rv);
    if (lane == 0) varOut[P] = rv * (1.0f / 256.0f);
}

// ---------------------------------------------------------------------------
// Fused: MLP GEMMs (WMMA) + gating + values GEMM (WMMA) + output LN.
// Block = 256 threads (8 waves), handles 64 pixels = 16 groups.
// MLP is run as 4 low-pressure passes (2 accumulators each) to avoid spills.
// ---------------------------------------------------------------------------
__global__ __launch_bounds__(256) void fused_kernel(
    const _Float16* __restrict__ normF, const _Float16* __restrict__ gmF,
    const float* __restrict__ varBuf,
    const uint32_t* __restrict__ dW1p, const uint32_t* __restrict__ rW1p,
    const uint32_t* __restrict__ vWp,
    const float* __restrict__ db1, const float* __restrict__ dW2, const float* __restrict__ db2,
    const float* __restrict__ rb1, const float* __restrict__ rW2, const float* __restrict__ rb2,
    const float* __restrict__ vb,  const float* __restrict__ ebias,
    const float* __restrict__ seed, const float* __restrict__ og, const float* __restrict__ ob,
    float* __restrict__ out) {

    __shared__ _Float16 joint_h[16 * 768];      // 24 KB, A-staging for MLP GEMMs
    __shared__ _Float16 wsum_h[16 * CCH];       //  8 KB, A-staging for values GEMM
    __shared__ float    cand_s[16 * CCH];       // 16 KB
    __shared__ float    drive_s[64], res_s[64], var_s[64];
    __shared__ float    w_s[64], us_s[16];
    __shared__ float    red1[256], red2[256], mv_s[32];

    const int tid  = threadIdx.x;
    const int lane = tid & 31;
    const int wid  = tid >> 5;
    const int wg   = blockIdx.x;
    const size_t pixBase = (size_t)wg * 64;
    const int arow  = lane & 15;
    const int ahalf = lane >> 4;

    if (tid < 64) {
        drive_s[tid] = 0.f;
        res_s[tid]   = 0.f;
        var_s[tid]   = varBuf[pixBase + tid];
    }
    __syncthreads();

    const uint32_t* joint_u = reinterpret_cast<const uint32_t*>(joint_h);

    // ---- two MLPs over 4 M-tiles of 16 pixels each ----
    for (int mt = 0; mt < 4; ++mt) {
        const size_t P0 = pixBase + mt * 16;
        for (int idx = tid; idx < 16 * 256; idx += 256) {
            int row = idx >> 8, c = idx & 255;
            size_t P = P0 + row;
            size_t gI = P >> 2;
            _Float16 n = normF[P * CCH + c];
            _Float16 g = gmF[gI * CCH + c];
            joint_h[row * 768 + c]       = n;
            joint_h[row * 768 + 256 + c] = g;
            joint_h[row * 768 + 512 + c] = (_Float16)((float)n * (float)g);
        }
        __syncthreads();

        // 4 passes: pass 0,1 -> drive MLP; pass 2,3 -> resistance MLP.
        // Each pass handles N-tiles {nt0, nt0+8} with only 2 accumulators.
        for (int pass = 0; pass < 4; ++pass) {
            const uint32_t* Wp = (pass < 2) ? dW1p : rW1p;
            const int nt0 = wid + (pass & 1) * 16;
            const int nt1 = nt0 + 8;

            v8f acc0, acc1;
            #pragma unroll
            for (int i = 0; i < 8; ++i) { acc0[i] = 0.f; acc1[i] = 0.f; }

            for (int kt = 0; kt < 24; ++kt) {
                AF a;
                const int base = arow * 384 + kt * 16 + ahalf * 8;
                #pragma unroll
                for (int j = 0; j < 4; ++j) { a.u[j] = joint_u[base + j]; a.u[4 + j] = joint_u[base + 4 + j]; }
                v16h b0 = *(const v16h*)(Wp + ((size_t)(kt * 32 + nt0) * 32 + lane) * 8);
                acc0 = __builtin_amdgcn_wmma_f32_16x16x32_f16(
                    false, a.v, false, b0, (short)0, acc0, false, false);
                v16h b1 = *(const v16h*)(Wp + ((size_t)(kt * 32 + nt1) * 32 + lane) * 8);
                acc1 = __builtin_amdgcn_wmma_f32_16x16x32_f16(
                    false, a.v, false, b1, (short)0, acc1, false, false);
            }

            // gelu + dot with W2 column, reduce over N (width-16 butterflies)
            const float* B1 = (pass < 2) ? db1 : rb1;
            const float* W2 = (pass < 2) ? dW2 : rW2;
            float* acc_s    = (pass < 2) ? drive_s : res_s;
            #pragma unroll
            for (int t = 0; t < 2; ++t) {
                const int nt = (t == 0) ? nt0 : nt1;
                const int N = nt * 16 + arow;
                const float b1v = B1[N];
                const float w2v = W2[N];
                #pragma unroll
                for (int r = 0; r < 8; ++r) {
                    float av = (t == 0) ? acc0[r] : acc1[r];
                    float p = gelu_exact(av + b1v) * w2v;
                    #pragma unroll
                    for (int m = 8; m >= 1; m >>= 1) p += __shfl_xor(p, m, 16);
                    if (arow == 0) {
                        atomicAdd(&acc_s[mt * 16 + ahalf * 8 + r], p);
                    }
                }
            }
        }
        __syncthreads();
    }

    // ---- gating: energy -> softmax with null logit 0 ----
    if (tid < 16) {
        float e[4];
        const float bb2 = db2[0], rr2 = rb2[0], eb = ebias[0];
        #pragma unroll
        for (int g = 0; g < 4; ++g) {
            int rr = tid * 4 + g;
            float drv = drive_s[rr] + bb2;
            float raw = res_s[rr] + rr2;
            float sp = (raw > 20.f) ? raw : log1pf(expf(raw));
            float resist = sp + var_s[rr];
            float en = drv / (resist + 1e-6f) + eb;
            e[g] = fminf(3.f, fmaxf(-3.f, en));
        }
        float mx = 0.f;
        #pragma unroll
        for (int g = 0; g < 4; ++g) mx = fmaxf(mx, e[g]);
        float nullw = expf(-mx);
        float den = nullw, ww[4];
        #pragma unroll
        for (int g = 0; g < 4; ++g) { ww[g] = expf(e[g] - mx); den += ww[g]; }
        float inv = 1.f / den;
        #pragma unroll
        for (int g = 0; g < 4; ++g) w_s[tid * 4 + g] = ww[g] * inv;
        us_s[tid] = 1.f - nullw * inv;
    }
    __syncthreads();

    // ---- weighted sum of normalized rows (candidate = wsum @ vW + us*vb) ----
    {
        int p = tid >> 4, seg = tid & 15;
        float w0 = w_s[p * 4], w1 = w_s[p * 4 + 1], w2w = w_s[p * 4 + 2], w3 = w_s[p * 4 + 3];
        size_t base0 = (pixBase + p * 4) * CCH;
        #pragma unroll
        for (int i = 0; i < 16; ++i) {
            int c = seg * 16 + i;
            float sv = w0 * (float)normF[base0 + c]
                     + w1 * (float)normF[base0 + CCH + c]
                     + w2w * (float)normF[base0 + 2 * CCH + c]
                     + w3 * (float)normF[base0 + 3 * CCH + c];
            wsum_h[p * CCH + c] = (_Float16)sv;
        }
    }
    __syncthreads();

    // ---- values GEMM: wsum(16x256) @ vW(256x256) via WMMA ----
    {
        v8f accv0, accv1;
        #pragma unroll
        for (int i = 0; i < 8; ++i) { accv0[i] = 0.f; accv1[i] = 0.f; }
        const uint32_t* wsum_u = reinterpret_cast<const uint32_t*>(wsum_h);
        for (int kt = 0; kt < 8; ++kt) {
            AF a;
            const int base = arow * 128 + kt * 16 + ahalf * 8;
            #pragma unroll
            for (int j = 0; j < 4; ++j) { a.u[j] = wsum_u[base + j]; a.u[4 + j] = wsum_u[base + 4 + j]; }
            v16h b0 = *(const v16h*)(vWp + ((size_t)(kt * 16 + wid) * 32 + lane) * 8);
            accv0 = __builtin_amdgcn_wmma_f32_16x16x32_f16(
                false, a.v, false, b0, (short)0, accv0, false, false);
            v16h b1 = *(const v16h*)(vWp + ((size_t)(kt * 16 + wid + 8) * 32 + lane) * 8);
            accv1 = __builtin_amdgcn_wmma_f32_16x16x32_f16(
                false, a.v, false, b1, (short)0, accv1, false, false);
        }
        #pragma unroll
        for (int t = 0; t < 2; ++t) {
            int N = (wid + t * 8) * 16 + arow;
            #pragma unroll
            for (int r = 0; r < 8; ++r) {
                int M = ahalf * 8 + r;
                cand_s[M * CCH + N] = (t == 0) ? accv0[r] : accv1[r];
            }
        }
    }
    __syncthreads();

    // ---- seed blend + output LayerNorm ----
    {
        int p = tid >> 4, seg = tid & 15;
        float us = us_s[p];
        float s1 = 0.f, s2 = 0.f;
        #pragma unroll
        for (int i = 0; i < 16; ++i) {
            int c = seg * 16 + i;
            float val = cand_s[p * CCH + c] + us * vb[c];
            float sd = seed[c];
            float nv = sd + us * (val - sd);
            cand_s[p * CCH + c] = nv;
            s1 += nv; s2 += nv * nv;
        }
        red1[tid] = s1; red2[tid] = s2;
    }
    __syncthreads();
    if (tid < 16) {
        float s1 = 0.f, s2 = 0.f;
        #pragma unroll
        for (int i = 0; i < 16; ++i) { s1 += red1[tid * 16 + i]; s2 += red2[tid * 16 + i]; }
        float m = s1 * (1.0f / 256.0f);
        float v = s2 * (1.0f / 256.0f) - m * m;
        mv_s[tid * 2] = m;
        mv_s[tid * 2 + 1] = rsqrtf(v + 1e-5f);
    }
    __syncthreads();
    {
        int p = tid >> 4, seg = tid & 15;
        float m = mv_s[p * 2], rs = mv_s[p * 2 + 1];
        size_t obase = ((size_t)wg * 16 + p) * CCH;
        #pragma unroll
        for (int i = 0; i < 16; ++i) {
            int c = seg * 16 + i;
            out[obase + c] = (cand_s[p * CCH + c] - m) * rs * og[c] + ob[c];
        }
    }
}

// ---------------------------------------------------------------------------
extern "C" void kernel_launch(void* const* d_in, const int* in_sizes, int n_in,
                              void* d_out, int out_size, void* d_ws, size_t ws_size,
                              hipStream_t stream) {
    const float* states = (const float*)d_in[0];
    const float* ln_c_g = (const float*)d_in[1];
    const float* ln_c_b = (const float*)d_in[2];
    const float* seed   = (const float*)d_in[3];
    const float* dW1    = (const float*)d_in[4];
    const float* db1    = (const float*)d_in[5];
    const float* dW2    = (const float*)d_in[6];
    const float* db2    = (const float*)d_in[7];
    const float* rW1    = (const float*)d_in[8];
    const float* rb1    = (const float*)d_in[9];
    const float* rW2    = (const float*)d_in[10];
    const float* rb2    = (const float*)d_in[11];
    const float* vW     = (const float*)d_in[12];
    const float* vb     = (const float*)d_in[13];
    const float* ebias  = (const float*)d_in[14];
    const float* og     = (const float*)d_in[15];
    const float* ob     = (const float*)d_in[16];

    uint8_t* ws = (uint8_t*)d_ws;
    _Float16* normF = (_Float16*)(ws + 0);            // 65536*256*2  = 33,554,432 B
    _Float16* gmF   = (_Float16*)(ws + 33554432);     // 16384*256*2  =  8,388,608 B
    float*    varB  = (float*)   (ws + 41943040);     // 65536*4      =    262,144 B
    uint32_t* dW1p  = (uint32_t*)(ws + 42205184);     // 24*32*32*32  =    786,432 B
    uint32_t* rW1p  = (uint32_t*)(ws + 42991616);     //                  786,432 B
    uint32_t* vWp   = (uint32_t*)(ws + 43778048);     //  8*16*32*32  =    131,072 B

    // pack weights into WMMA B-fragment layout (f16)
    pack_b_kernel<<<96, 256, 0, stream>>>(dW1, HID, 32, dW1p, 24 * 32 * 32);
    pack_b_kernel<<<96, 256, 0, stream>>>(rW1, HID, 32, rW1p, 24 * 32 * 32);
    pack_b_kernel<<<16, 256, 0, stream>>>(vW, CCH, 16, vWp, 8 * 16 * 32);

    // stage 1: LN + group stats
    stage1_kernel<<<NGRP, 128, 0, stream>>>(states, ln_c_g, ln_c_b, normF, gmF, varB);

    // fused MLP/gating/values/output-LN
    fused_kernel<<<NPIX / 64, 256, 0, stream>>>(normF, gmF, varB, dW1p, rW1p, vWp,
                                                db1, dW2, db2, rb1, rW2, rb2,
                                                vb, ebias, seed, og, ob, (float*)d_out);
}
